// LSTMModel_11295763989098
// MI455X (gfx1250) — matrix operand order repaired and checked by reading, not compiled, not run on hardware
//
#include <hip/hip_runtime.h>
#include <stdint.h>

// ---------------------------------------------------------------------------
// 5-layer LSTM (H=50, B=4096, T=256) for gfx1250 using v_wmma_f32_16x16x32_f16
// 64 WGs x 8 waves (2 waves/SIMD32 for WMMA/VALU co-execution), Btile=64
// ---------------------------------------------------------------------------
#define Hh   50
#define Bsz  4096
#define Tt   256
#define BT   64            // batch tile per workgroup
#define NWG  (Bsz / BT)    // 64 workgroups
#define NTH  256           // 8 waves of 32 -> 2 waves per SIMD32

// padded GEMM shape per step: M=64, N=256 (4 gates x 64), K=128 (h:0..63, in:64..127)
#define KP      128
#define NP      256
#define ASTR    136        // A LDS row stride in f16 (272B, 16B aligned, bank-rotating)
#define WSTR    264        // W LDS row stride in f16 (528B, 16B aligned, bank-rotating)
#define YSTR    64         // global y row stride (f16), 128B rows
#define SMEM_BYTES ((KP*WSTR + BT*ASTR) * 2 + NP * 4)

typedef _Float16 v8h  __attribute__((ext_vector_type(8)));
typedef _Float16 v16h __attribute__((ext_vector_type(16)));
typedef float    v8f  __attribute__((ext_vector_type(8)));

__device__ __forceinline__ float sigm(float x) {
    return 1.0f / (1.0f + __expf(-x));
}
__device__ __forceinline__ float tanh_f(float x) {
    return 1.0f - 2.0f / (__expf(2.0f * x) + 1.0f);
}

__device__ __forceinline__ v16h ld_frag(const _Float16* p0, const _Float16* p1) {
    v8h lo = *(const v8h*)p0;
    v8h hi = *(const v8h*)p1;
    return __builtin_shufflevector(lo, hi, 0,1,2,3,4,5,6,7,8,9,10,11,12,13,14,15);
}

// cooperative load of the input-side of A (cols 64..127) for timestep t
__device__ __forceinline__ void load_input(int layer, int t, int tid, int b0,
                                           const float* x, const _Float16* yg,
                                           _Float16* Ash) {
    if (layer == 0) {
        if (tid < BT) {
            float xv = x[(size_t)(b0 + tid) * Tt + t];  // x: [B][T][1]
            Ash[tid * ASTR + 64] = (_Float16)xv;        // cols 65..127 stay zero
        }
    } else {
        int row = tid >> 2;          // 0..63
        int seg = tid & 3;           // 4 x 16 f16 = 64 f16 per row
        const uint4* src = (const uint4*)(yg + ((size_t)t * Bsz + b0 + row) * YSTR + seg * 16);
        uint4* dst = (uint4*)(Ash + row * ASTR + 64 + seg * 16);
        dst[0] = src[0];
        dst[1] = src[1];
    }
}

__global__ void __launch_bounds__(NTH, 1)
lstm_layer_kernel(int layer,
                  const float* __restrict__ x,
                  const float* __restrict__ w_ih0,   // [200][1]
                  const float* __restrict__ w_hh0,   // [200][50]
                  const float* __restrict__ b0v,     // [200]
                  const float* __restrict__ w_ih,    // [4][200][50]
                  const float* __restrict__ w_hh,    // [4][200][50]
                  const float* __restrict__ bv,      // [4][200]
                  _Float16* __restrict__ yg,         // [T][B][64] f16 (in-place)
                  float* __restrict__ hTg)           // [B][64] f32
{
    extern __shared__ char smem[];
    _Float16* Wsh  = (_Float16*)smem;                       // [KP][WSTR]
    _Float16* Ash  = Wsh + KP * WSTR;                       // [BT][ASTR]
    float*    Bias = (float*)(smem + (KP * WSTR + BT * ASTR) * 2); // [NP]

    const int tid  = threadIdx.x;
    const int b0   = blockIdx.x * BT;
    const int lane = tid & 31;
    const int w    = tid >> 5;        // wave 0..7
    const int mt   = w >> 1;          // M-tile 0..3  (rows mt*16..mt*16+15)
    const int p    = w & 1;           // j-group (tiles {2p,2p+1} within each gate)
    const int nl   = lane & 15;
    const int hiL  = lane >> 4;

    // ---- build padded f16 weight matrix B[k][n] in LDS --------------------
    for (int idx = tid; idx < KP * NP; idx += NTH) {
        int k = idx >> 8;            // 0..127
        int n = idx & 255;           // 0..255
        int g = n >> 6, j = n & 63;
        float v = 0.0f;
        if (j < Hh) {
            int unit = g * Hh + j;
            if (layer == 0) {
                if (k < Hh)       v = w_hh0[unit * Hh + k];
                else if (k == 64) v = w_ih0[unit];
            } else {
                int lb = (layer - 1) * 4 * Hh * Hh;  // 10000 per layer
                if (k < Hh)                      v = w_hh[lb + unit * Hh + k];
                else if (k >= 64 && k < 64 + Hh) v = w_ih[lb + unit * Hh + (k - 64)];
            }
        }
        Wsh[k * WSTR + n] = (_Float16)v;
    }
    for (int n = tid; n < NP; n += NTH) {
        int g = n >> 6, j = n & 63;
        float v = 0.0f;
        if (j < Hh) v = (layer == 0) ? b0v[g * Hh + j] : bv[(layer - 1) * 4 * Hh + g * Hh + j];
        Bias[n] = v;
    }
    // zero A (h0 = 0, and all pad columns)
    for (int idx = tid; idx < BT * ASTR; idx += NTH) Ash[idx] = (_Float16)0.0f;
    __syncthreads();
    load_input(layer, 0, tid, b0, x, yg, Ash);

    // bias is t-invariant: keep this wave's 8 column-bias values in registers
    float bias_r[4][2];
#pragma unroll
    for (int g = 0; g < 4; ++g)
#pragma unroll
        for (int u = 0; u < 2; ++u)
            bias_r[g][u] = Bias[(g * 4 + 2 * p + u) * 16 + nl];

    float creg[2][8];
#pragma unroll
    for (int u = 0; u < 2; ++u)
#pragma unroll
        for (int v = 0; v < 8; ++v) creg[u][v] = 0.0f;

    __syncthreads();

#pragma unroll 1
    for (int t = 0; t < Tt; ++t) {
        // prefetch next timestep's input rows while the GEMM runs
        if (layer != 0 && t + 1 < Tt) {
            int row = tid >> 2, seg = tid & 3;
            __builtin_prefetch(yg + ((size_t)(t + 1) * Bsz + b0 + row) * YSTR + seg * 16, 0, 3);
        }

        // ---- GEMM: acc[g][u] = bias + A(64x128) * B(128x16-tile) ----------
        v8f acc[4][2];
#pragma unroll
        for (int g = 0; g < 4; ++g)
#pragma unroll
            for (int u = 0; u < 2; ++u) {
                float bi = bias_r[g][u];
                v8f a = {bi, bi, bi, bi, bi, bi, bi, bi};
                acc[g][u] = a;
            }

#pragma unroll
        for (int kt = 0; kt < KP / 32; ++kt) {
            const _Float16* ap = Ash + (mt * 16 + nl) * ASTR + kt * 32 + hiL * 8;
            v16h af = ld_frag(ap, ap + 16);
#pragma unroll
            for (int g = 0; g < 4; ++g)
#pragma unroll
                for (int u = 0; u < 2; ++u) {
                    int nt = g * 4 + 2 * p + u;
                    const _Float16* bp = Wsh + (kt * 32 + hiL * 8) * WSTR + nt * 16 + nl;
                    v16h bf;
#pragma unroll
                    for (int s = 0; s < 16; ++s)
                        bf[s] = bp[((s & 7) + ((s >> 3) << 4)) * WSTR];
                    acc[g][u] = __builtin_amdgcn_wmma_f32_16x16x32_f16(
                        false, af, false, bf, (short)0, acc[g][u], false, false);
                }
        }

        __syncthreads();   // all A reads of this step complete

        // ---- gates fused in registers; write h -> A, y -> global ----------
#pragma unroll
        for (int u = 0; u < 2; ++u) {
            int q = 2 * p + u;
#pragma unroll
            for (int v = 0; v < 8; ++v) {
                float iv = acc[0][u][v];
                float fv = acc[1][u][v];
                float gv = acc[2][u][v];
                float ov = acc[3][u][v];
                float cN = sigm(fv) * creg[u][v] + sigm(iv) * tanh_f(gv);
                float hN = sigm(ov) * tanh_f(cN);
                creg[u][v] = cN;

                int row = mt * 16 + v + 8 * hiL;
                int j   = q * 16 + nl;
                if (j < Hh) Ash[row * ASTR + j] = (_Float16)hN;
                _Float16 yv = (j < Hh) ? (_Float16)hN : (_Float16)0.0f;
                yg[(size_t)t * Bsz * YSTR + (size_t)(b0 + row) * YSTR + j] = yv;
                if (t == Tt - 1 && j < Hh) hTg[(b0 + row) * YSTR + j] = hN;
            }
        }
        if (t + 1 < Tt) load_input(layer, t + 1, tid, b0, x, yg, Ash);

        __syncthreads();   // A ready for next step
    }
}

__global__ void __launch_bounds__(256, 1)
fc_kernel(const float* __restrict__ hT,
          const float* __restrict__ fc_w,
          const float* __restrict__ fc_b,
          float* __restrict__ out) {
    int b = blockIdx.x * blockDim.x + threadIdx.x;
    if (b < Bsz) {
        float s = fc_b[0];
#pragma unroll
        for (int j = 0; j < Hh; ++j) s += hT[b * YSTR + j] * fc_w[j];
        out[b] = s;
    }
}

extern "C" void kernel_launch(void* const* d_in, const int* in_sizes, int n_in,
                              void* d_out, int out_size, void* d_ws, size_t ws_size,
                              hipStream_t stream) {
    const float* x     = (const float*)d_in[0];
    const float* w_ih0 = (const float*)d_in[1];
    const float* w_hh0 = (const float*)d_in[2];
    const float* b0v   = (const float*)d_in[3];
    const float* w_ih  = (const float*)d_in[4];
    const float* w_hh  = (const float*)d_in[5];
    const float* bv    = (const float*)d_in[6];
    const float* fc_w  = (const float*)d_in[7];
    const float* fc_b  = (const float*)d_in[8];
    float* out = (float*)d_out;

    // workspace: y[T][B][64] f16 (134.2 MB, L2-resident), then hT[B][64] f32
    _Float16* yg = (_Float16*)d_ws;
    float* hTg = (float*)((char*)d_ws + (size_t)Tt * Bsz * YSTR * sizeof(_Float16));

    for (int l = 0; l < 5; ++l) {
        lstm_layer_kernel<<<NWG, NTH, SMEM_BYTES, stream>>>(
            l, x, w_ih0, w_hh0, b0v, w_ih, w_hh, bv, yg, hTg);
    }
    fc_kernel<<<(Bsz + 255) / 256, 256, 0, stream>>>(hTg, fc_w, fc_b, out);
}
